// QuantizedLinear_18184891532044
// MI455X (gfx1250) — 8-bit matrix operand order repaired and checked by reading, compiled, not run on hardware
//
#include <hip/hip_runtime.h>


typedef int   v8i __attribute__((ext_vector_type(8)));
typedef float v4f __attribute__((ext_vector_type(4)));
typedef int   v4i __attribute__((ext_vector_type(4)));

#define M_TOTAL 8192
#define N_TOTAL 4096
#define K_TOTAL 4096
#define KT_CNT  (K_TOTAL / 64)
#define A_BYTES ((size_t)M_TOTAL * K_TOTAL)

__global__ __launch_bounds__(256)
void quantize_x_swizzle(const float* __restrict__ x,
                        const float* __restrict__ act_scale,
                        unsigned int* __restrict__ a8) {
    const unsigned int d = blockIdx.x * 256 + threadIdx.x;
    const float inv = 1.0f / act_scale[0];

    const unsigned int tile = d >> 8;
    const unsigned int rem  = d & 255;
    const unsigned int lane = rem >> 3;
    const unsigned int j    = rem & 7;
    const unsigned int mt   = tile >> 6;
    const unsigned int kt   = tile & 63;

    const unsigned int m = (mt << 4) + (lane & 15);
    const unsigned int k = (kt << 6) + ((lane >> 4) << 3) + ((j >> 1) << 4) + ((j & 1) << 2);

    const v4f* __restrict__ x4 = (const v4f*)x;
    const v4f v = __builtin_nontemporal_load(&x4[m * (K_TOTAL / 4) + (k >> 2)]);

    const int b0 = (int)fminf(fmaxf(rintf(v.x * inv), -128.0f), 127.0f);
    const int b1 = (int)fminf(fmaxf(rintf(v.y * inv), -128.0f), 127.0f);
    const int b2 = (int)fminf(fmaxf(rintf(v.z * inv), -128.0f), 127.0f);
    const int b3 = (int)fminf(fmaxf(rintf(v.w * inv), -128.0f), 127.0f);

    a8[d] = (unsigned int)((b0 & 255) | ((b1 & 255) << 8) |
                           ((b2 & 255) << 16) | ((b3 & 255) << 24));
}

__global__ __launch_bounds__(256)
void pack_w_swizzle(const int* __restrict__ w,
                    unsigned int* __restrict__ b8) {
    const unsigned int d    = blockIdx.x * 256 + threadIdx.x;
    const unsigned int tile = d >> 8;
    const unsigned int rem  = d & 255;
    const unsigned int lane = rem >> 3;
    const unsigned int j    = rem & 7;
    const unsigned int nt   = tile >> 6;
    const unsigned int kt   = tile & 63;

    const unsigned int n = (nt << 4) + (lane & 15);
    const unsigned int k = (kt << 6) + ((lane >> 4) << 3) + ((j >> 1) << 4) + ((j & 1) << 2);

    const v4i* __restrict__ w4 = (const v4i*)w;
    const v4i v = __builtin_nontemporal_load(&w4[n * (K_TOTAL / 4) + (k >> 2)]);

    b8[d] = (unsigned int)((v.x & 255) | ((v.y & 255) << 8) |
                           ((v.z & 255) << 16) | ((v.w & 255) << 24));
}

__global__ __launch_bounds__(256)
void gemm_wmma_iu8(const v8i* __restrict__ a8,
                   const v8i* __restrict__ b8,
                   const float* __restrict__ wscale,
                   const float* __restrict__ ascale,
                   const float* __restrict__ bias,
                   float* __restrict__ out) {
    const unsigned int lane  = threadIdx.x & 31;
    const unsigned int wave  = threadIdx.x >> 5;
    const unsigned int waveM = wave & 1;
    const unsigned int waveN = wave >> 1;

    const unsigned int mTile0 = blockIdx.y * 8  + waveM * 4;
    const unsigned int nTile0 = blockIdx.x * 16 + waveN * 4;

    const v8i* aP = a8 + ((size_t)(mTile0 << 11) + lane);
    const v8i* bP = b8 + ((size_t)(nTile0 << 11) + lane);

    v8i acc[4][4] = {};

    for (int c = 0; c < KT_CNT / 4; ++c) {
#pragma unroll
        for (int u = 0; u < 4; ++u) {
            v8i af[4], bf[4];
#pragma unroll
            for (int i = 0; i < 4; ++i)
                af[i] = aP[i * 2048 + u * 32];
#pragma unroll
            for (int jn = 0; jn < 4; ++jn)
                bf[jn] = bP[jn * 2048 + u * 32];
#pragma unroll
            for (int i = 0; i < 4; ++i)
#pragma unroll
                for (int jn = 0; jn < 4; ++jn)
                    acc[i][jn] = __builtin_amdgcn_wmma_i32_16x16x64_iu8(
                         true, af[i],  true, bf[jn],
                        acc[i][jn],  false,  false);
        }
        aP += 4 * 32;
        bP += 4 * 32;
    }

    const float scale = wscale[0] * ascale[0];
    const unsigned int col  = lane & 15;
    const unsigned int rsel = (lane >> 4) << 3;

#pragma unroll
    for (int i = 0; i < 4; ++i) {
#pragma unroll
        for (int jn = 0; jn < 4; ++jn) {
            const unsigned int n  = ((nTile0 + jn) << 4) + col;
            const float bv = bias[n];
            const unsigned int m0 = ((mTile0 + i) << 4) + rsel;
#pragma unroll
            for (int r = 0; r < 8; ++r) {
                const float val = (float)acc[i][jn][r] * scale + bv;
                __builtin_nontemporal_store(val, &out[(m0 + r) * N_TOTAL + n]);
            }
        }
    }
}

extern "C" void kernel_launch(void* const* d_in, const int* in_sizes, int n_in,
                              void* d_out, int out_size, void* d_ws, size_t ws_size,
                              hipStream_t stream) {
    const float* x    = (const float*)d_in[0];
    const int*   qw   = (const int*)d_in[1];
    const float* wsc  = (const float*)d_in[2];
    const float* asc  = (const float*)d_in[3];
    const float* bias = (const float*)d_in[4];
    float*       out  = (float*)d_out;

    unsigned char* a8 = (unsigned char*)d_ws;
    unsigned char* b8 = a8 + A_BYTES;

    quantize_x_swizzle<<<(M_TOTAL * (K_TOTAL / 4)) / 256, 256, 0, stream>>>(
        x, asc, (unsigned int*)a8);
    pack_w_swizzle<<<(N_TOTAL * (K_TOTAL / 4)) / 256, 256, 0, stream>>>(
        qw, (unsigned int*)b8);
    dim3 grid(N_TOTAL / 256, M_TOTAL / 128);
    gemm_wmma_iu8<<<grid, 256, 0, stream>>>((const v8i*)a8, (const v8i*)b8,
                                            wsc, asc, bias, out);
}
